// LocalAttention_65927747993815
// MI455X (gfx1250) — matrix operand order repaired and checked by reading, not compiled, not run on hardware
//
#include <hip/hip_runtime.h>

// ---------------------------------------------------------------------------
// Local (windowed, causal, look_backward=1) attention with rotary embeddings.
// B*H = 64, T = 4096, E = 64, WINDOW = 128 -> 2048 independent (bh, window)
// tiles. One workgroup (256 threads = 8 wave32) per tile.
//   GEMM-1: S = rot(Q)(128x64) * rot(K)^T(64x256)   via v_wmma_f32_16x16x32_f16
//   softmax rows (causal + pad mask)
//   GEMM-2: O = P(128x256) * V(256x64)              via v_wmma_f32_16x16x32_f16
// ---------------------------------------------------------------------------

typedef _Float16 f16x16 __attribute__((ext_vector_type(16)));
typedef _Float16 f16x8  __attribute__((ext_vector_type(8)));
typedef _Float16 f16x4  __attribute__((ext_vector_type(4)));
typedef float    f32x8  __attribute__((ext_vector_type(8)));

#define WIN     128
#define EDIM    64
#define TLEN    4096
#define NWIN    (TLEN / WIN)      // 32 windows
#define KW      (2 * WIN)         // 256 keys per window (backward + current)
#define QPITCH  72                // halves per Q row (64 + 8 pad), 144B = 9*16
#define KPITCH  72
#define VPITCH  264               // halves per Vt row (256 + 8 pad), 528B = 33*16
#define PPITCH  264

#define VT_HALVES (EDIM * VPITCH)          // 16896
#define QS_HALVES (WIN * QPITCH)           //  9216
#define KS_HALVES (KW * KPITCH)            // 18432
#define PS_HALVES (8 * 16 * PPITCH)        // 33792 (8 waves x 16 rows)
// Ps aliases the Qs+Ks region (Q/K dead after GEMM-1); Vt persists.
#define OVL_HALVES ((PS_HALVES > (QS_HALVES + KS_HALVES)) ? PS_HALVES : (QS_HALVES + KS_HALVES))
#define SMEM_BYTES ((VT_HALVES + OVL_HALVES) * 2)   // = 101376 B (~99 KB)

__global__ __launch_bounds__(256) void local_attn_rotary_wmma(
    const float* __restrict__ q, const float* __restrict__ k,
    const float* __restrict__ v, float* __restrict__ out)
{
    extern __shared__ __align__(16) _Float16 smem[];
    _Float16* Vt = smem;                       // [EDIM][VPITCH]  (transposed V)
    _Float16* Qs = smem + VT_HALVES;           // [WIN][QPITCH]   (phase 1)
    _Float16* Ks = Qs + QS_HALVES;             // [KW][KPITCH]    (phase 1)
    _Float16* Ps = smem + VT_HALVES;           // [8][16][PPITCH] (phase 2, alias)

    const int blk = blockIdx.x;
    const int w   = blk & (NWIN - 1);
    const int bh  = blk >> 5;                  // log2(NWIN) == 5
    const int tid = threadIdx.x;

    const size_t base = (size_t)bh * TLEN * EDIM;
    const float* qg = q + base + (size_t)w * WIN * EDIM;
    const int ktok0 = (w - 1) * WIN;           // token index of extended-window key 0
    const float LOG1E4 = 9.210340371976184f;   // ln(10000)

    // ---------------- Fill Q (rotary, f32 -> f16) ----------------
    for (int task = tid; task < WIN * 8; task += 256) {
        const int row = task >> 3;
        const int c   = (task & 7) << 2;       // pair base e in [0,31], step 4
        float4 x1 = *(const float4*)(qg + row * EDIM + c);
        float4 x2 = *(const float4*)(qg + row * EDIM + c + 32);
        const float t = (float)(w * WIN + row);
        _Float16 lo[4], hi2[4];
        #pragma unroll
        for (int j = 0; j < 4; ++j) {
            const float fr = __expf(-(float)(c + j) * (1.0f / 32.0f) * LOG1E4);
            float sn, cs; __sincosf(t * fr, &sn, &cs);
            const float a = ((const float*)&x1)[j];
            const float b = ((const float*)&x2)[j];
            lo[j]  = (_Float16)(a * cs - b * sn);
            hi2[j] = (_Float16)(b * cs + a * sn);
        }
        *(f16x4*)&Qs[row * QPITCH + c]      = *(f16x4*)lo;
        *(f16x4*)&Qs[row * QPITCH + c + 32] = *(f16x4*)hi2;
    }

    // ---------------- Fill K (rotary, f32 -> f16) ----------------
    for (int task = tid; task < KW * 8; task += 256) {
        const int row = task >> 3;
        const int c   = (task & 7) << 2;
        int tok = ktok0 + row;
        if (tok < 0) tok = 0;                  // first window's pad keys: masked later
        const float* kg = k + base + (size_t)tok * EDIM;
        float4 x1 = *(const float4*)(kg + c);
        float4 x2 = *(const float4*)(kg + c + 32);
        const float t = (float)tok;
        _Float16 lo[4], hi2[4];
        #pragma unroll
        for (int j = 0; j < 4; ++j) {
            const float fr = __expf(-(float)(c + j) * (1.0f / 32.0f) * LOG1E4);
            float sn, cs; __sincosf(t * fr, &sn, &cs);
            const float a = ((const float*)&x1)[j];
            const float b = ((const float*)&x2)[j];
            lo[j]  = (_Float16)(a * cs - b * sn);
            hi2[j] = (_Float16)(b * cs + a * sn);
        }
        *(f16x4*)&Ks[row * KPITCH + c]      = *(f16x4*)lo;
        *(f16x4*)&Ks[row * KPITCH + c + 32] = *(f16x4*)hi2;
    }

    // ---------------- Fill Vt (transposed V, f16) ----------------
    for (int task = tid; task < KW * 16; task += 256) {
        const int row = task >> 4;
        const int c   = (task & 15) << 2;
        int tok = ktok0 + row;
        if (tok < 0) tok = 0;                  // pad rows get prob 0 anyway
        float4 x = *(const float4*)(v + base + (size_t)tok * EDIM + c);
        #pragma unroll
        for (int j = 0; j < 4; ++j)
            Vt[(c + j) * VPITCH + row] = (_Float16)(((const float*)&x)[j]);
    }
    __syncthreads();

    // ---------------- GEMM-1: S = Q * K^T (per-wave 16 query rows) -----------
    const int lane  = tid & 31;
    const int wv    = tid >> 5;
    const int mbase = wv * 16;
    const int lc    = lane & 15;
    const int hi    = lane >> 4;

    f16x16 aq[2];
    #pragma unroll
    for (int kt = 0; kt < 2; ++kt) {           // A frag: lane=row, split-K halves
        const int e0 = kt * 32 + hi * 8;
        union { f16x16 f; f16x8 h[2]; } u;
        u.h[0] = *(const f16x8*)&Qs[(mbase + lc) * QPITCH + e0];
        u.h[1] = *(const f16x8*)&Qs[(mbase + lc) * QPITCH + e0 + 16];
        aq[kt] = u.f;
    }

    f32x8 s[16];
    #pragma unroll
    for (int nt = 0; nt < 16; ++nt) {
        f32x8 acc = {};
        #pragma unroll
        for (int kt = 0; kt < 2; ++kt) {       // B frag: lane=column, split-K halves
            const int e0 = kt * 32 + hi * 8;
            union { f16x16 f; f16x8 h[2]; } u;
            u.h[0] = *(const f16x8*)&Ks[(nt * 16 + lc) * KPITCH + e0];
            u.h[1] = *(const f16x8*)&Ks[(nt * 16 + lc) * KPITCH + e0 + 16];
            acc = __builtin_amdgcn_wmma_f32_16x16x32_f16(
                false, aq[kt], false, u.f, (short)0, acc, false, false);
        }
        const int jcol = nt * 16 + lc;         // key index 0..255
        #pragma unroll
        for (int r = 0; r < 8; ++r) {
            const int i = mbase + hi * 8 + r;  // query row 0..127
            const float val = acc[r] * 0.125f; // E^-0.5
            const bool dead = (jcol > i + WIN) || (w == 0 && jcol < WIN);
            acc[r] = dead ? -1.0e30f : val;
        }
        s[nt] = acc;
    }

    // ---------------- softmax over 256 keys per row ----------------
    f32x8 mx = s[0];
    #pragma unroll
    for (int nt = 1; nt < 16; ++nt)
        #pragma unroll
        for (int r = 0; r < 8; ++r) mx[r] = fmaxf(mx[r], s[nt][r]);
    #pragma unroll
    for (int off = 1; off < 16; off <<= 1)     // stays inside the 16-lane half
        #pragma unroll
        for (int r = 0; r < 8; ++r) mx[r] = fmaxf(mx[r], __shfl_xor(mx[r], off, 32));

    f32x8 sum = {};
    #pragma unroll
    for (int nt = 0; nt < 16; ++nt)
        #pragma unroll
        for (int r = 0; r < 8; ++r) {
            const float e = __expf(s[nt][r] - mx[r]);
            s[nt][r] = e; sum[r] += e;
        }
    #pragma unroll
    for (int off = 1; off < 16; off <<= 1)
        #pragma unroll
        for (int r = 0; r < 8; ++r) sum[r] += __shfl_xor(sum[r], off, 32);

    f32x8 inv;
    #pragma unroll
    for (int r = 0; r < 8; ++r) inv[r] = 1.0f / sum[r];

    __syncthreads();                           // Q/K region dead -> reuse as Ps

    // ---------------- P -> LDS (row-major per wave, f16) ----------------
    _Float16* PsW = Ps + wv * 16 * PPITCH;
    #pragma unroll
    for (int nt = 0; nt < 16; ++nt)
        #pragma unroll
        for (int r = 0; r < 8; ++r)
            PsW[(hi * 8 + r) * PPITCH + nt * 16 + lc] =
                (_Float16)(s[nt][r] * inv[r]);
    __syncthreads();

    // ---------------- GEMM-2: O = P * V ----------------
    f32x8 o[4] = {};
    #pragma unroll
    for (int kt = 0; kt < 8; ++kt) {
        union { f16x16 f; f16x8 h[2]; } ua;    // A frag from P (lane=row)
        const int j0 = kt * 32 + hi * 8;
        ua.h[0] = *(const f16x8*)&PsW[lc * PPITCH + j0];
        ua.h[1] = *(const f16x8*)&PsW[lc * PPITCH + j0 + 16];
        const int jb = kt * 32 + hi * 8;
        #pragma unroll
        for (int nt = 0; nt < 4; ++nt) {       // B frag from Vt (lane=column e)
            union { f16x16 f; f16x8 h[2]; } ub;
            ub.h[0] = *(const f16x8*)&Vt[(nt * 16 + lc) * VPITCH + jb];
            ub.h[1] = *(const f16x8*)&Vt[(nt * 16 + lc) * VPITCH + jb + 16];
            o[nt] = __builtin_amdgcn_wmma_f32_16x16x32_f16(
                false, ua.f, false, ub.f, (short)0, o[nt], false, false);
        }
    }

    // ---------------- store O (f32, coalesced across lanes) ----------------
    float* og = out + base + (size_t)w * WIN * EDIM;
    #pragma unroll
    for (int nt = 0; nt < 4; ++nt)
        #pragma unroll
        for (int r = 0; r < 8; ++r)
            og[(size_t)(mbase + hi * 8 + r) * EDIM + nt * 16 + lc] = o[nt][r];
}

extern "C" void kernel_launch(void* const* d_in, const int* in_sizes, int n_in,
                              void* d_out, int out_size, void* d_ws, size_t ws_size,
                              hipStream_t stream) {
    (void)n_in; (void)out_size; (void)d_ws; (void)ws_size;
    const float* q = (const float*)d_in[0];
    const float* k = (const float*)d_in[1];
    const float* v = (const float*)d_in[2];
    float* out = (float*)d_out;
    const int BH = in_sizes[0] / (TLEN * EDIM);   // B*H = 64
    dim3 grid(BH * NWIN);                          // 2048 workgroups
    dim3 block(256);                               // 8 wave32
    local_attn_rotary_wmma<<<grid, block, SMEM_BYTES, stream>>>(q, k, v, out);
}
